// LSTM_28656021799670
// MI455X (gfx1250) — compile-verified, matrix operand order checked by reading, not run on hardware
//
#include <hip/hip_runtime.h>

// ---------------------------------------------------------------------------
// 2-layer LSTM for MI455X (gfx1250), bf16 WMMA + fp32 accumulate.
// Persistent kernel per layer (32 WGs x 256 thr), grid barrier per timestep.
// Per step each WG computes 64 rows x 32 hidden cols: gates GEMM over
// K=2048 ([x_t | h_{t-1}] vs [W_ih | W_hh] bf16, L2-resident) fused with the
// LSTM cell applied in-register on the WMMA accumulators.
// A-panel staging via the Tensor Data Mover (tensor_load_to_lds +
// s_wait_tensorcnt, 6-arg clang-23 builtin form) -- wave 0 issues two 128KB
// 1D-tile descriptors with TDM padding (16B per 1024B -> 2080B row pitch);
// ASYNCcnt global_load_async_to_lds fallback kept.
// ---------------------------------------------------------------------------

typedef __attribute__((ext_vector_type(16))) __bf16          bf16x16;
typedef __attribute__((ext_vector_type(8)))  __bf16          bf16x8;
typedef __attribute__((ext_vector_type(8)))  float           f32x8;

union BFrag { bf16x8 h[2]; bf16x16 v; };

#define T_STEPS 512
#define BATCH   64
#define HID     1024
#define KTOT    2048          // I + H
#define NWG     32            // 1024 cols / 32 cols per WG

// TDM-padded LDS layout: data gets 16B pad after every 1024B (pad_interval=7,
// pad_amount=3). A 1024-element bf16 half-row (2048B) = 2 segments -> pitch
// 2080B. Block 0 = x_t (64x1024), block 1 = h_{t-1} (64x1024), each
// 64*2080 = 133120 B.
#define LDS_BLOCK   133120
#define SMEM_BYTES  (2 * LDS_BLOCK)       // 266240 B < 320KB/WGP

// byte offset of element (row, col) of the A panel (col 0..2047)
__device__ __forceinline__ unsigned lds_off(int row, int col) {
    unsigned blk = (unsigned)col >> 10;          // 0 = x_t, 1 = h
    unsigned cb  = (unsigned)col & 1023u;
    return blk * (unsigned)LDS_BLOCK + (unsigned)row * 2080u
         + cb * 2u + ((cb >> 9) << 4);           // 16B pad after 512 elements
}

__device__ __forceinline__ unsigned short f2bf(float f) {
    unsigned u = __builtin_bit_cast(unsigned, f);
    u += 0x7FFFu + ((u >> 16) & 1u);             // round-to-nearest-even
    return (unsigned short)(u >> 16);
}

// Branch-free sigmoid on native v_exp_f32 (2^x) + v_rcp_f32.
__device__ __forceinline__ float sigm(float x) {
    return __builtin_amdgcn_rcpf(1.0f + __builtin_amdgcn_exp2f(x * -1.44269504088896340736f));
}

// Native CDNA5 v_tanh_f32 (TRANS op).
__device__ __forceinline__ float fast_tanh(float x) {
#if __has_builtin(__builtin_amdgcn_tanhf)
    return __builtin_amdgcn_tanhf(x);
#else
    float y;
    asm volatile("v_tanh_f32 %0, %1\n\tv_nop" : "=v"(y) : "v"(x));
    return y;
#endif
}

// Issue one TDM descriptor: contiguous 128KB global -> LDS at lds_base with
// 16B-per-1024B padding. D# layout per CDNA5 ISA ch.8: group0 = {count=1,
// lds_addr, global_addr[56:0], type=2}; group1 = {data_size=8B, pad_enable,
// pad_interval=7 (256 dw), pad_amount=3 (4 dw), tensor_dim0=tile_dim0=16384,
// tile_dim1=1}. Remaining groups zero (dims 2..4 unused).
#if __has_builtin(__builtin_amdgcn_tensor_load_to_lds)
#define HAVE_TDM 1
typedef __attribute__((ext_vector_type(4))) unsigned tdm_v4u;
typedef __attribute__((ext_vector_type(8))) int      tdm_v8i;
typedef __attribute__((ext_vector_type(4))) int      tdm_v4i;
__device__ __forceinline__ void tdm_load_128k(unsigned lds_base, const void* gsrc) {
    unsigned long long ga = (unsigned long long)gsrc;
    tdm_v4u g0 = { 1u,                                   // count=1 valid D#
                   lds_base,                             // lds_addr
                   (unsigned)ga,                         // global_addr[31:0]
                   ((unsigned)(ga >> 32) & 0x01FFFFFFu) | 0x80000000u }; // [56:32]|type=2
    tdm_v8i g1 = { 0x07D30000,   // data_size=3(8B), pad_en, pad_int=7, pad_amt=3
                   0x40000000,   // tensor_dim0[15:0]=16384 in bits[31:16]
                   0x00010000,   // tensor_dim0 hi=0, tensor_dim1[15:0]=1
                   0x40000000,   // tensor_dim1 hi=0, tile_dim0=16384
                   1,            // tile_dim1=1, tile_dim2=0
                   16384,        // tensor_dim0_stride lo
                   0, 0 };
    tdm_v4i z4 = { 0, 0, 0, 0 };
    tdm_v8i z8 = { 0, 0, 0, 0, 0, 0, 0, 0 };
    // amdgpu-toolchain (clang-23) 6-arg form:
    // (v4u g0, v8i g1, v4i, v4i, v8i, i32 cpol)
    __builtin_amdgcn_tensor_load_to_lds(g0, g1, z4, z4, z8, 0);
}
#else
#define HAVE_TDM 0
#endif

// ---------------------------- prep kernels ---------------------------------

__global__ void k_prep_weights(const float* __restrict__ w_ih,
                               const float* __restrict__ w_hh,
                               const float* __restrict__ b_ih,
                               const float* __restrict__ b_hh,
                               unsigned short* __restrict__ wcat,
                               float* __restrict__ bc) {
    const long long total = 2LL * 4096 * 2048;
    long long stride = (long long)gridDim.x * blockDim.x;
    for (long long i = (long long)blockIdx.x * blockDim.x + threadIdx.x;
         i < total; i += stride) {
        int l   = (int)(i >> 23);                 // 4096*2048 = 2^23
        int rem = (int)(i & ((1 << 23) - 1));
        int n   = rem >> 11;
        int k   = rem & 2047;
        float v = (k < 1024) ? w_ih[(((long long)l * 4096 + n) << 10) + k]
                             : w_hh[(((long long)l * 4096 + n) << 10) + (k - 1024)];
        wcat[i] = f2bf(v);
    }
    int gid = blockIdx.x * blockDim.x + threadIdx.x;
    if (gid < 2 * 4096) bc[gid] = b_ih[gid] + b_hh[gid];
}

__global__ void k_prep_x(const float* __restrict__ x,
                         unsigned short* __restrict__ xb) {
    const long long total = (long long)T_STEPS * BATCH * 1024;
    long long stride = (long long)gridDim.x * blockDim.x;
    for (long long i = (long long)blockIdx.x * blockDim.x + threadIdx.x;
         i < total; i += stride)
        xb[i] = f2bf(x[i]);
}

__global__ void k_init_layer(const float* __restrict__ h0,
                             const float* __restrict__ c0, int l,
                             unsigned short* __restrict__ hbuf,
                             float* __restrict__ cst,
                             unsigned* __restrict__ counter) {
    int gid = blockIdx.x * blockDim.x + threadIdx.x;
    if (gid < BATCH * HID) {
        hbuf[gid] = f2bf(h0[l * BATCH * HID + gid]);
        cst[gid]  = c0[l * BATCH * HID + gid];
    }
    if (gid == 0) *counter = 0u;
}

// ------------------------- persistent LSTM kernel --------------------------

extern __shared__ unsigned short sA[];   // TDM-padded A panel (see lds_off)

__global__ __launch_bounds__(256, 1)
void lstm_persist(const unsigned short* __restrict__ xin,   // (T,64,1024) bf16
                  const unsigned short* __restrict__ wcat,  // (4096,2048) bf16
                  const float* __restrict__ bc,             // (4096) fp32
                  unsigned short* __restrict__ hping,       // (64,1024) bf16
                  unsigned short* __restrict__ hpong,       // (64,1024) bf16
                  float* __restrict__ cstate,               // (64,1024) fp32
                  float* __restrict__ out_f32,              // (T,64,1024) or null
                  unsigned short* __restrict__ xnext,       // (T,64,1024) bf16 or null
                  float* __restrict__ hT_out,               // (64,1024) fp32
                  float* __restrict__ cT_out,               // (64,1024) fp32
                  unsigned* __restrict__ counter) {
    const int tid   = threadIdx.x;
    const int wave  = tid >> 5;
    const int lane  = tid & 31;
    const int r     = wave & 3;         // row tile 0..3 (rows 16r..16r+15)
    const int cs    = wave >> 2;        // col slice 0..1 (16 cols each)
    const int n0    = blockIdx.x * 32;  // this WG owns hidden cols n0..n0+31
    const int l15   = lane & 15;
    const int lhalf = (lane >> 4) << 3; // 0 or 8 (K-offset per WMMA layout)

    // Per-lane 32-bit element offsets from the uniform wcat base + bias/gate.
    const int ncol = n0 + cs * 16 + l15;     // hidden column this lane handles
    unsigned wofs[4];
    float bias[4];
#pragma unroll
    for (int g = 0; g < 4; ++g) {
        wofs[g] = (unsigned)(g * 1024 + ncol) * KTOT + (unsigned)lhalf;
        bias[g] = bc[g * 1024 + ncol];
    }

    const char* sAb = (const char*)sA;

    for (int t = 0; t < T_STEPS; ++t) {
        const unsigned short* hread  = (t & 1) ? hpong : hping;
        unsigned short*       hwrite = (t & 1) ? hping : hpong;

        // ---- stage A = [x_t | h_{t-1}] (2 x 128KB contiguous) into LDS ----
        const unsigned short* xt = xin + (size_t)t * BATCH * 1024;
#if HAVE_TDM
        if (wave == 0) {                       // one wave drives the DMA
            tdm_load_128k(0u,                  xt);
            tdm_load_128k((unsigned)LDS_BLOCK, hread);
            __builtin_amdgcn_s_wait_tensorcnt((short)0);
        }
        __syncthreads();
#else
        for (int i = tid; i < BATCH * (KTOT / 8); i += 256) {   // 16384 chunks
            int row = i >> 8;
            int col = (i & 255) * 8;           // element col 0..2047
            const unsigned short* src = (col < 1024)
                ? (xt    + row * 1024 + col)
                : (hread + row * 1024 + (col - 1024));
            unsigned ldsoff = lds_off(row, col);
            asm volatile("global_load_async_to_lds_b128 %0, %1, off"
                         :: "v"(ldsoff), "v"(src) : "memory");
        }
        asm volatile("s_wait_asynccnt 0" ::: "memory");
        __syncthreads();
#endif

        // ---- gates GEMM: 4 gate tiles (16x16) per wave, K = 2048 ----------
        f32x8 acc[4] = {};
        const int am = 16 * r + l15;           // A row this lane supplies
        for (int kk = 0; kk < KTOT; kk += 32) {
            BFrag a;
            a.h[0] = *reinterpret_cast<const bf16x8*>(sAb + lds_off(am, kk + lhalf));
            a.h[1] = *reinterpret_cast<const bf16x8*>(sAb + lds_off(am, kk + lhalf + 16));
#pragma unroll
            for (int g = 0; g < 4; ++g) {
                BFrag b;
                const unsigned short* bp = wcat + (wofs[g] + (unsigned)kk);
                b.h[0] = *reinterpret_cast<const bf16x8*>(bp);
                b.h[1] = *reinterpret_cast<const bf16x8*>(bp + 16);
                acc[g] = __builtin_amdgcn_wmma_f32_16x16x32_bf16(
                    false, a.v, false, b.v, (short)0, acc[g], false, false);
            }
        }

        // ---- fused LSTM cell on accumulators (C layout: lane=N, vgpr=M) ---
#pragma unroll
        for (int v = 0; v < 8; ++v) {
            int m = 16 * r + v + lhalf;               // batch row 0..63
            size_t idx = (size_t)m * HID + ncol;
            float ig = acc[0][v] + bias[0];
            float fg = acc[1][v] + bias[1];
            float gg = acc[2][v] + bias[2];
            float og = acc[3][v] + bias[3];
            float cp = cstate[idx];
            float cn = sigm(fg) * cp + sigm(ig) * fast_tanh(gg);
            float h  = sigm(og) * fast_tanh(cn);
            cstate[idx] = cn;
            hwrite[idx] = f2bf(h);
            size_t oidx = (size_t)t * BATCH * HID + idx;
            if (out_f32) out_f32[oidx] = h;
            if (xnext)   xnext[oidx]   = f2bf(h);
            if (t == T_STEPS - 1) { hT_out[idx] = h; cT_out[idx] = cn; }
        }

        // ---- grid-wide barrier before next timestep -----------------------
        __threadfence();
        __syncthreads();
        if (tid == 0) {
            __hip_atomic_fetch_add(counter, 1u, __ATOMIC_RELEASE,
                                   __HIP_MEMORY_SCOPE_AGENT);
            unsigned target = (unsigned)gridDim.x * (unsigned)(t + 1);
            while (__hip_atomic_load(counter, __ATOMIC_ACQUIRE,
                                     __HIP_MEMORY_SCOPE_AGENT) < target)
                __builtin_amdgcn_s_sleep(2);
        }
        __syncthreads();
        __threadfence();
    }
}

// ------------------------------ launcher -----------------------------------

extern "C" void kernel_launch(void* const* d_in, const int* in_sizes, int n_in,
                              void* d_out, int out_size, void* d_ws, size_t ws_size,
                              hipStream_t stream) {
    const float* x    = (const float*)d_in[0];
    const float* h0   = (const float*)d_in[1];
    const float* c0   = (const float*)d_in[2];
    const float* w_ih = (const float*)d_in[3];
    const float* w_hh = (const float*)d_in[4];
    const float* b_ih = (const float*)d_in[5];
    const float* b_hh = (const float*)d_in[6];

    float* out  = (float*)d_out;                              // (T,B,H)
    float* hfin = out + (size_t)T_STEPS * BATCH * HID;        // (L,B,H)
    float* cfin = hfin + 2LL * BATCH * HID;                   // (L,B,H)

    // workspace layout (all offsets 128B aligned)
    char* ws = (char*)d_ws;
    const size_t WCAT_SZ = 2LL * 4096 * 2048 * 2;             // 33.5 MB bf16
    const size_t BC_SZ   = 2LL * 4096 * 4;
    const size_t XB_SZ   = (size_t)T_STEPS * BATCH * 1024 * 2; // 67 MB bf16
    size_t off = 0;
    unsigned short* wcat = (unsigned short*)(ws + off); off += WCAT_SZ;
    float*          bc   = (float*)(ws + off);          off += BC_SZ;
    unsigned short* xb0  = (unsigned short*)(ws + off); off += XB_SZ;
    unsigned short* xb1  = (unsigned short*)(ws + off); off += XB_SZ;
    unsigned short* hp0  = (unsigned short*)(ws + off); off += (size_t)BATCH * HID * 2 + 128;
    unsigned short* hp1  = (unsigned short*)(ws + off); off += (size_t)BATCH * HID * 2 + 128;
    float*          cst  = (float*)(ws + off);          off += (size_t)BATCH * HID * 4;
    unsigned*       cnt  = (unsigned*)(ws + off);       off += 256;

    // one-time converts: weights+bias -> bf16 concat [W_ih|W_hh], x -> bf16
    k_prep_weights<<<dim3(4096), dim3(256), 0, stream>>>(w_ih, w_hh, b_ih, b_hh,
                                                         wcat, bc);
    k_prep_x<<<dim3(4096), dim3(256), 0, stream>>>(x, xb0);

    for (int l = 0; l < 2; ++l) {
        k_init_layer<<<dim3(256), dim3(256), 0, stream>>>(h0, c0, l, hp0, cst, cnt);
        const unsigned short* xin  = (l == 0) ? xb0 : xb1;
        float*          outp  = (l == 1) ? out : nullptr;
        unsigned short* xnxt  = (l == 0) ? xb1 : nullptr;
        lstm_persist<<<dim3(NWG), dim3(256), SMEM_BYTES, stream>>>(
            xin, wcat + (size_t)l * 4096 * 2048, bc + (size_t)l * 4096,
            hp0, hp1, cst, outp, xnxt,
            hfin + (size_t)l * BATCH * HID, cfin + (size_t)l * BATCH * HID, cnt);
    }
}
